// STDPLocallyConnected_1675037245661
// MI455X (gfx1250) — matrix operand order repaired and checked by reading, not compiled, not run on hardware
//
#include <hip/hip_runtime.h>
#include <hip/hip_bf16.h>

// ---------------------------------------------------------------------------
// STDP locally-connected layer for MI455X (gfx1250).
//   out[b,m,i,j] = sum_{c,x,y} spikes[b,c,refl(i+x-3),refl(j+y-3)] * W[i,j,c,x,y,m]
//   mem = membrane*0.99 + out ; spike = mem>1 ; mem clamped to 1 on spike
// Per pixel: 16x16 GEMM, K=784 (padded 800) -> 25x v_wmma_f32_16x16x32_f16.
// Weights (205.5 MB, single-pass stream) are DMA'd by the Tensor Data Mover
// into per-wave double-buffered LDS; spike tiles staged once per block in LDS
// as f16 (spikes are exact 0/1). A-fragment addressing goes through a
// precomputed k->LDS-offset table whose padded entries point at a zeroed LDS
// region, so the inner loop has no divisions and no guard selects.
// ---------------------------------------------------------------------------

typedef __attribute__((ext_vector_type(16))) _Float16     v16h;
typedef __attribute__((ext_vector_type(8)))  float        v8f;
typedef __attribute__((ext_vector_type(4)))  unsigned int v4u;
typedef __attribute__((ext_vector_type(8)))  int          v8i;
typedef __attribute__((ext_vector_type(4)))  int          v4i;

#define B_      16
#define C_      16
#define H_      64
#define W_      64
#define KS      7
#define PADR    3
#define CKK     784                     // C_ * KS * KS
#define KPAD    800                     // next multiple of 32
#define PIXW    4                       // pixels per block (one per wave)
#define NWAVE   4
#define TCOLS   (PIXW + 2 * PADR)       // 10
#define CHUNK   128                     // K rows per TDM chunk
#define NFULL   6                       // 6*128 = 768, tail = 16 rows
#define WBUF_FLOATS (CHUNK * C_)        // 2048 floats = 8 KB
#define WBUF_BYTES  (WBUF_FLOATS * 4)
#define SPIKE_HALFS (C_ * KS * TCOLS * B_)   // 17920
#define SPIKE_BYTES (SPIKE_HALFS * 2)        // 35840
#define ZPAD_BYTES  SPIKE_BYTES              // 128 zeroed halfs
#define ZPAD_SIZE   256
#define OFFTAB_BYTES (ZPAD_BYTES + ZPAD_SIZE)     // u16[800] @ 36096
#define WB_BASE     (OFFTAB_BYTES + KPAD * 2)     // 37696 (64B aligned)
#define LDS_BYTES   (WB_BASE + NWAVE * 2 * WBUF_BYTES)  // 103232

__device__ __forceinline__ int refl(int p) {
  p = (p < 0) ? -p : p;
  return (p > 63) ? (126 - p) : p;
}

// One TDM descriptor: contiguous 1-row 2D tile of `nelem` f32, global -> LDS
// byte offset lds_off (dynamic LDS assumed to start at 0). D# packing per
// CDNA5 ISA 8.3/8.4 (count=1, type=2, data_size=4B). cpol=NT: single-pass
// weight stream should not thrash L2.
__device__ __forceinline__ void tdm_load_1d(unsigned lds_off, const float* g,
                                            unsigned nelem) {
  unsigned long long ga = (unsigned long long)(uintptr_t)g;
  v4u g0;
  g0[0] = 1u;                                   // count=1, is_restore=0
  g0[1] = lds_off;                              // lds_addr
  g0[2] = (unsigned)ga;                         // global_addr[31:0]
  g0[3] = (unsigned)((ga >> 32) & 0x1FFFFFFu)   // global_addr[56:32]
        | (2u << 30);                           // type=2
  v8i g1;
  g1[0] = (int)(2u << 16);                      // wg_mask=0, data_size=4B
  g1[1] = (int)((nelem & 0xFFFFu) << 16);       // tensor_dim0[15:0]
  g1[2] = (int)(((nelem >> 16) & 0xFFFFu)       // tensor_dim0[31:16]
        | (1u << 16));                          // tensor_dim1 = 1
  g1[3] = (int)((nelem & 0xFFFFu) << 16);       // tile_dim0 = nelem (<=2048)
  g1[4] = 1;                                    // tile_dim1=1, tile_dim2=0
  g1[5] = (int)nelem;                           // tensor_dim0_stride[31:0]
  g1[6] = (int)((nelem & 0xFFFFu) << 16);       // dim1_stride lo
  g1[7] = (int)(nelem >> 16);                   // dim1_stride hi
  v4i z4 = {0, 0, 0, 0};
  v8i z8 = {0, 0, 0, 0, 0, 0, 0, 0};
  __builtin_amdgcn_tensor_load_to_lds(g0, g1, z4, z4, z8, /*cpol NT*/ 1);
}

template <int N>
__device__ __forceinline__ void wait_tensorcnt() {
#if __has_builtin(__builtin_amdgcn_s_wait_tensorcnt)
  __builtin_amdgcn_s_wait_tensorcnt(N);
#else
  asm volatile("s_wait_tensorcnt %0" :: "i"(N) : "memory");
#endif
  asm volatile("" ::: "memory");  // keep LDS reads below the wait
}

// One 16x16x32 fragment: A from spike tile via offset table (zero-padded k),
// B from TDM weight buffer (f32 -> f16 cvt). All table/B reads use
// compile-time ds immediate offsets off a single base per fragment.
__device__ __forceinline__ v8f frag_mma(v8f acc, const char* smem,
                                        const unsigned short* offTab,
                                        const float* wb, int kGlob0, int kLoc0,
                                        int laneOffBytes, int bIdx, int laneHi) {
  v16h a, bf;
  const unsigned short* tb = offTab + kGlob0 + laneHi * 8;
  const float* wbl = wb + (kLoc0 + laneHi * 8) * C_ + bIdx;
#pragma unroll
  for (int t = 0; t < 16; ++t) {
    const int kp = (t & 8) ? (t + 8) : t;  // A K-pattern (16-bit 16x32 layout)
    unsigned off = tb[kp];                 // ds_load_u16, imm offset
    a[t]  = *(const _Float16*)(smem + off + laneOffBytes);
    bf[t] = (_Float16)wbl[kp * C_];        // ds_load_b32, imm offset + cvt
  }
  return __builtin_amdgcn_wmma_f32_16x16x32_f16(false, a, false, bf,
                                                (short)0, acc, false, false);
}

__global__ void __launch_bounds__(128)
stdp_lc_wmma_kernel(const float* __restrict__ in_spikes,
                    const float* __restrict__ weights,
                    const float* __restrict__ membrane,
                    float* __restrict__ out /* [spike | mem] */) {
  extern __shared__ char smem[];
  _Float16*       spikeT = (_Float16*)smem;               // [c][x(7)][col(10)][b]
  _Float16*       zpad   = (_Float16*)(smem + ZPAD_BYTES);
  unsigned short* offTab = (unsigned short*)(smem + OFFTAB_BYTES);

  const int tid    = threadIdx.x;
  const int lane   = tid & 31;
  const int laneHi = lane >> 4;         // 0: lanes 0-15, 1: lanes 16-31
  const int bIdx   = lane & 15;         // A row (batch) / B col (out-chan)
  const int wave   = tid >> 5;          // 0..3, one pixel column per wave
  const int i  = blockIdx.y;            // output row
  const int j0 = blockIdx.x * PIXW;     // first output col of this block
  const int j  = j0 + wave;             // this wave's output col
  const int laneOffBytes = (wave * B_ + bIdx) * 2;  // (jj*16 + b) halfs

  // ---- Stage reflected spike tile (f16; spikes are exact 0/1) -------------
  for (int idx = tid; idx < SPIKE_HALFS; idx += 128) {   // 140 * 128 exact
    int b   = idx & 15;
    int col = (idx >> 4) % TCOLS;
    int r   = (idx / (16 * TCOLS)) % KS;
    int c   = idx / (16 * TCOLS * KS);
    int h   = refl(i + r - PADR);
    int w   = refl(j0 + col - PADR);
    float s = in_spikes[(((size_t)b * C_ + c) * H_ + h) * W_ + w];
    spikeT[(((size_t)c * KS + r) * TCOLS + col) * B_ + b] = (_Float16)s;
  }
  zpad[tid] = (_Float16)0.0f;  // 128 threads x 1 = whole pad region
  // k -> spike-tile byte offset table; padded k points at the zero region.
  for (int k = tid; k < KPAD; k += 128) {
    unsigned short off;
    if (k < CKK) {
      int c = k / 49, rm = k - c * 49;
      int x = rm / 7, y = rm - x * 7;
      off = (unsigned short)((((c * KS + x) * TCOLS) + y) * B_ * 2);
    } else {
      off = (unsigned short)ZPAD_BYTES;
    }
    offTab[k] = off;
  }
  __syncthreads();

  // ---- Stream this pixel's weight block (784x16 f32) via TDM --------------
  const float* wpix = weights + (size_t)(i * W_ + j) * (CKK * C_);
  const unsigned wb0 = WB_BASE + (unsigned)(wave * 2) * WBUF_BYTES;
  const unsigned wb1 = wb0 + WBUF_BYTES;

  v8f acc = {};
  tdm_load_1d(wb0, wpix, CHUNK * C_);   // prologue: chunk 0 -> buf 0

  for (int ch = 0; ch < NFULL; ++ch) {
    const int rows = (ch == NFULL - 1) ? (CKK - NFULL * CHUNK) : CHUNK;
    tdm_load_1d(((ch & 1) ? wb0 : wb1),
                wpix + (size_t)(ch + 1) * CHUNK * C_, (unsigned)(rows * C_));
    wait_tensorcnt<1>();                 // in-order TDM: current buffer ready
    const float* wb = (const float*)(smem + ((ch & 1) ? wb1 : wb0));
#pragma unroll
    for (int f = 0; f < 4; ++f)
      acc = frag_mma(acc, smem, offTab, wb, ch * CHUNK + f * 32, f * 32,
                     laneOffBytes, bIdx, laneHi);
  }
  // Tail: k 768..799. A is zero for k>=784 via the table; B rows 16..31 of
  // buf0 hold stale-but-finite chunk-4 weights -> contribute exactly 0.
  wait_tensorcnt<0>();
  acc = frag_mma(acc, smem, offTab, (const float*)(smem + wb0), NFULL * CHUNK,
                 0, laneOffBytes, bIdx, laneHi);

  // ---- LIF epilogue. D layout: VGPR r -> M(batch) = r + laneHi*8 ----------
  const int m = bIdx;  // N = out channel
  const size_t CHWS = (size_t)C_ * H_ * W_;  // batch stride (65536 elems)
  const size_t base = (((size_t)(laneHi * 8) * C_ + m) * H_ + i) * W_ + j;
  const float* mb = membrane + base;
  float* os = out + base;
  float* om = out + (size_t)B_ * CHWS + base;
#pragma unroll
  for (int r = 0; r < 8; ++r) {
    float mv = mb[r * CHWS] * 0.99f + acc[r];
    bool sp = mv > 1.0f;
    os[r * CHWS] = sp ? 1.0f : 0.0f;
    om[r * CHWS] = sp ? 1.0f : mv;
  }
}

extern "C" void kernel_launch(void* const* d_in, const int* in_sizes, int n_in,
                              void* d_out, int out_size, void* d_ws, size_t ws_size,
                              hipStream_t stream) {
  const float* in_spikes = (const float*)d_in[0];
  const float* weights   = (const float*)d_in[1];
  const float* membrane  = (const float*)d_in[2];
  float* out = (float*)d_out;

  (void)in_sizes; (void)n_in; (void)out_size; (void)d_ws; (void)ws_size;

  // ~101 KB dynamic LDS > default cap; opt in (deterministic, capture-safe).
  (void)hipFuncSetAttribute((const void*)stdp_lc_wmma_kernel,
                            hipFuncAttributeMaxDynamicSharedMemorySize,
                            LDS_BYTES);

  dim3 grid(W_ / PIXW, H_);   // 16 x 64 blocks
  dim3 block(NWAVE * 32);     // 4 waves, one pixel each
  stdp_lc_wmma_kernel<<<grid, block, LDS_BYTES, stream>>>(
      in_spikes, weights, membrane, out);
}
